// AggreGCN_4836133175935
// MI455X (gfx1250) — matrix operand order repaired and checked by reading, not compiled, not run on hardware
//
#include <hip/hip_runtime.h>
#include <hip/hip_bf16.h>

typedef __attribute__((ext_vector_type(16))) _Float16 v16h;
typedef __attribute__((ext_vector_type(8)))  _Float16 v8h;
typedef __attribute__((ext_vector_type(8)))  float    v8f;

#define NNODES 100000
#define NEDGES 1600000
#define NFEATC 256
#define NHIDC  128
#define NCLSC  40
#define NCLSP  48      // padded class count (3 x 16 tiles)

// ---------------------------------------------------------------- utilities
__global__ void zero_f4(float4* __restrict__ p, int n4) {
    int i = blockIdx.x * blockDim.x + threadIdx.x;
    if (i < n4) p[i] = make_float4(0.f, 0.f, 0.f, 0.f);
}

__global__ void deg_kernel(const int* __restrict__ dst, float* __restrict__ deg) {
    int i = blockIdx.x * blockDim.x + threadIdx.x;
    if (i < NEDGES) atomicAdd(&deg[dst[i]], 1.0f);
}

// deg -> D^{-1/2} in place; +1.0 accounts for the self loop, so deg>0 always.
__global__ void dinv_kernel(float* __restrict__ deg) {
    int i = blockIdx.x * blockDim.x + threadIdx.x;
    if (i < NNODES) deg[i] = rsqrtf(deg[i] + 1.0f);
}

// Transpose+convert weights: Wt[n][k] (f16) = W[k][n] (f32); zero-pad n >= Nreal.
__global__ void conv_wt(const float* __restrict__ W, _Float16* __restrict__ Wt,
                        int K, int Nreal, int total /* = Npad*K */) {
    int idx = blockIdx.x * blockDim.x + threadIdx.x;
    if (idx >= total) return;
    int nn = idx / K, k = idx - nn * K;
    Wt[idx] = (nn < Nreal) ? (_Float16)W[k * Nreal + nn] : (_Float16)0.f;
}

// ------------------------------------------------------- WMMA f16/f32 GEMM
// One wave computes one 16x16 tile of C = A(f32 row-major, lda=KT) @ Wt^T.
// Wt is f16 column-major [Npad][KT] so every B fragment is 2 contiguous b128s.
// All strides compile-time -> immediate-offset loads, f32 accumulation.
template<int KT, int NT, int LDC, bool BIAS, bool RELU>
__global__ __launch_bounds__(32) void wmma_gemm(
    const float* __restrict__ A, const _Float16* __restrict__ Bt,
    const float* __restrict__ bias, float* __restrict__ C)
{
    const int lane = threadIdx.x;
    const int m0   = blockIdx.x << 4;
    const int n0   = blockIdx.y << 4;
    const int sub  = lane & 15;      // row within A-frag / col within B,C-frag
    const int hi   = lane >> 4;      // lane-half selector
    const int n    = n0 + sub;       // output column for this lane

    const float*    arow = A  + (size_t)(m0 + sub) * KT;
    const _Float16* brow = Bt + (size_t)n * KT;

    v8f c = {};
#pragma unroll
    for (int k0 = 0; k0 < KT; k0 += 32) {
        v16h a, b;
        // A frag (16x32 f16): lane holds row m0+sub, K in [k0+8hi,+8) U [k0+16+8hi,+8)
#pragma unroll
        for (int e = 0; e < 8; ++e) {
            a[e]     = (_Float16)arow[k0 + 8 * hi + e];
            a[8 + e] = (_Float16)arow[k0 + 16 + 8 * hi + e];
        }
        // B frag (32x16 f16): lane holds column n, K in [k0+8hi,+8) U [k0+16+8hi,+8)
        v8h blo = *(const v8h*)(brow + k0 + 8 * hi);
        v8h bhi = *(const v8h*)(brow + k0 + 16 + 8 * hi);
#pragma unroll
        for (int e = 0; e < 8; ++e) { b[e] = blo[e]; b[8 + e] = bhi[e]; }
        c = __builtin_amdgcn_wmma_f32_16x16x32_f16(false, a, false, b,
                                                   (short)0, c, false, false);
    }

    const bool nok = (NT % 16 == 0) || (n < NT);
    float bv = (BIAS && nok) ? bias[n] : 0.f;
#pragma unroll
    for (int r = 0; r < 8; ++r) {        // C frag: row = m0 + r + 8*hi, col = n
        float v = c[r] + bv;
        if (RELU) v = fmaxf(v, 0.f);
        if (nok) C[(size_t)(m0 + r + 8 * hi) * LDC + n] = v;
    }
}

// ------------------------------------------------------- graph scatter ops
// acc[i,:] = h[i,:] * dinv[i]^2  (self-loop term initializes the accumulator)
__global__ void init_self_h(const float4* __restrict__ h, const float* __restrict__ dinv,
                            float4* __restrict__ acc) {
    int gid = blockIdx.x * blockDim.x + threadIdx.x;    // over N * 32 float4
    if (gid >= NNODES * (NHIDC / 4)) return;
    float di = dinv[gid >> 5];
    float w = di * di;
    float4 v = h[gid];
    acc[gid] = make_float4(v.x * w, v.y * w, v.z * w, v.w * w);
}

// one wave32 per edge: float4 gather of h[src], 4 f32 atomics into acc[dst]
__global__ void scatter_h(const int* __restrict__ src, const int* __restrict__ dst,
                          const float* __restrict__ dinv, const float* __restrict__ h,
                          float* __restrict__ acc) {
    int e    = (blockIdx.x * blockDim.x + threadIdx.x) >> 5;
    int lane = threadIdx.x & 31;
    if (e >= NEDGES) return;
    int s = src[e], d = dst[e];
    float w = dinv[s] * dinv[d];
    float4 hv = ((const float4*)(h + (size_t)s * NHIDC))[lane];
    float* o = acc + (size_t)d * NHIDC + lane * 4;
    atomicAdd(o + 0, hv.x * w);
    atomicAdd(o + 1, hv.y * w);
    atomicAdd(o + 2, hv.z * w);
    atomicAdd(o + 3, hv.w * w);
}

// x = relu(acc + b1) in place (float4)
__global__ void finalize_hidden(float4* __restrict__ x, const float4* __restrict__ b) {
    int gid = blockIdx.x * blockDim.x + threadIdx.x;    // over N * 32 float4
    if (gid >= NNODES * (NHIDC / 4)) return;
    float4 v = x[gid];
    float4 bb = b[gid & 31];
    v.x = fmaxf(v.x + bb.x, 0.f);
    v.y = fmaxf(v.y + bb.y, 0.f);
    v.z = fmaxf(v.z + bb.z, 0.f);
    v.w = fmaxf(v.w + bb.w, 0.f);
    x[gid] = v;
}

// out[i,:] = h2[i,:] * dinv[i]^2  (self-loop init, 10 float4 per row)
__global__ void init_self_cls(const float4* __restrict__ h2, const float* __restrict__ dinv,
                              float4* __restrict__ out) {
    int gid = blockIdx.x * blockDim.x + threadIdx.x;    // over N * 10 float4
    if (gid >= NNODES * (NCLSC / 4)) return;
    float di = dinv[gid / (NCLSC / 4)];
    float w = di * di;
    float4 v = h2[gid];
    out[gid] = make_float4(v.x * w, v.y * w, v.z * w, v.w * w);
}

__global__ void scatter_cls(const int* __restrict__ src, const int* __restrict__ dst,
                            const float* __restrict__ dinv, const float* __restrict__ h,
                            float* __restrict__ out) {
    int gid = blockIdx.x * blockDim.x + threadIdx.x;    // E*40 < 2^31
    if (gid >= NEDGES * NCLSC) return;
    int e = gid / NCLSC, c = gid - e * NCLSC;
    int s = src[e], d = dst[e];
    float w = dinv[s] * dinv[d];
    atomicAdd(&out[(size_t)d * NCLSC + c], h[(size_t)s * NCLSC + c] * w);
}

// one wave32 per node row: add bias, then log_softmax in place
__global__ void finalize_logsoftmax(const float* __restrict__ b2, float* __restrict__ out) {
    int row  = blockIdx.x * (blockDim.x >> 5) + (threadIdx.x >> 5);
    int lane = threadIdx.x & 31;
    if (row >= NNODES) return;                           // wave-uniform exit
    float* orow = out + (size_t)row * NCLSC;
    int c0 = lane, c1 = lane + 32;
    float x0 = 0.f, x1 = 0.f, v0 = -1e30f, v1 = -1e30f;
    if (c0 < NCLSC) { x0 = orow[c0] + b2[c0]; v0 = x0; }
    if (c1 < NCLSC) { x1 = orow[c1] + b2[c1]; v1 = x1; }
    float m = fmaxf(v0, v1);
#pragma unroll
    for (int off = 16; off > 0; off >>= 1) m = fmaxf(m, __shfl_xor(m, off, 32));
    float s = 0.f;
    if (c0 < NCLSC) s += __expf(x0 - m);
    if (c1 < NCLSC) s += __expf(x1 - m);
#pragma unroll
    for (int off = 16; off > 0; off >>= 1) s += __shfl_xor(s, off, 32);
    float lse = m + __logf(s);
    if (c0 < NCLSC) orow[c0] = x0 - lse;
    if (c1 < NCLSC) orow[c1] = x1 - lse;
}

// ---------------------------------------------------------------- launcher
extern "C" void kernel_launch(void* const* d_in, const int* in_sizes, int n_in,
                              void* d_out, int out_size, void* d_ws, size_t ws_size,
                              hipStream_t stream) {
    (void)in_sizes; (void)n_in; (void)out_size; (void)ws_size;
    const float* X  = (const float*)d_in[0];
    const int*   ei = (const int*)d_in[1];
    const float* W0 = (const float*)d_in[2];
    const float* b0 = (const float*)d_in[3];
    const float* W1 = (const float*)d_in[4];
    const float* b1 = (const float*)d_in[5];
    const float* W2 = (const float*)d_in[6];
    const float* b2 = (const float*)d_in[7];
    float* out = (float*)d_out;
    const int* src = ei;            // edge_index[0]
    const int* dst = ei + NEDGES;   // edge_index[1]

    // workspace: dinv | B0 (x0/x1, f32 NxNHID) | B1 (h1/h2, f32 NxNHID) | f16 weights
    char* ws = (char*)d_ws;
    float*     dinv = (float*)ws;
    float*     B0   = (float*)(ws + 400128);
    float*     B1   = (float*)(ws + 400128 + (size_t)NNODES * NHIDC * 4);
    _Float16*  W0t  = (_Float16*)(ws + 400128 + 2 * (size_t)NNODES * NHIDC * 4);
    _Float16*  W1t  = W0t + NHIDC * NFEATC;
    _Float16*  W2t  = W1t + NHIDC * NHIDC;
    // total: 400128 + 102,400,000 + 110,592 bytes =~ 102.9 MB

    // --- weight prep (tiny): f16 transposed, class dim padded to 48
    conv_wt<<<(NHIDC * NFEATC + 255) / 256, 256, 0, stream>>>(W0, W0t, NFEATC, NHIDC, NHIDC * NFEATC);
    conv_wt<<<(NHIDC * NHIDC + 255) / 256, 256, 0, stream>>>(W1, W1t, NHIDC, NHIDC, NHIDC * NHIDC);
    conv_wt<<<(NCLSP * NHIDC + 255) / 256, 256, 0, stream>>>(W2, W2t, NHIDC, NCLSC, NCLSP * NHIDC);

    // --- normalization weights
    zero_f4<<<(NNODES / 4 + 255) / 256, 256, 0, stream>>>((float4*)dinv, NNODES / 4);
    deg_kernel<<<(NEDGES + 255) / 256, 256, 0, stream>>>(dst, dinv);
    dinv_kernel<<<(NNODES + 255) / 256, 256, 0, stream>>>(dinv);

    // --- layer 0: x0 = relu(X @ W0 + b0)
    wmma_gemm<NFEATC, NHIDC, NHIDC, true, true>
        <<<dim3(NNODES / 16, NHIDC / 16), 32, 0, stream>>>(X, W0t, b0, B0);

    // --- hidden conv: h1 = x0 @ W1; acc = h1*dinv^2 + edge scatter; x1 = relu(acc+b1)
    wmma_gemm<NHIDC, NHIDC, NHIDC, false, false>
        <<<dim3(NNODES / 16, NHIDC / 16), 32, 0, stream>>>(B0, W1t, nullptr, B1);
    init_self_h<<<(NNODES * (NHIDC / 4) + 255) / 256, 256, 0, stream>>>
        ((const float4*)B1, dinv, (float4*)B0);
    scatter_h<<<(NEDGES * 32 + 255) / 256, 256, 0, stream>>>(src, dst, dinv, B1, B0);
    finalize_hidden<<<(NNODES * (NHIDC / 4) + 255) / 256, 256, 0, stream>>>
        ((float4*)B0, (const float4*)b1);

    // --- output conv: h2 = x1 @ W2 (ldc = 40, padded B, masked store)
    wmma_gemm<NHIDC, NCLSC, NCLSC, false, false>
        <<<dim3(NNODES / 16, NCLSP / 16), 32, 0, stream>>>(B0, W2t, nullptr, B1);
    init_self_cls<<<(NNODES * (NCLSC / 4) + 255) / 256, 256, 0, stream>>>
        ((const float4*)B1, dinv, (float4*)out);
    scatter_cls<<<(NEDGES * NCLSC + 255) / 256, 256, 0, stream>>>(src, dst, dinv, B1, out);
    finalize_logsoftmax<<<(NNODES + 7) / 8, 256, 0, stream>>>(b2, out);
}
